// MultiHeadAttention_43645457662018
// MI455X (gfx1250) — matrix operand order repaired and checked by reading, not compiled, not run on hardware
//
#include <hip/hip_runtime.h>
#include <hip/hip_bf16.h>

// ---------------------------------------------------------------------------
// Problem constants (from reference): B=2, S=2048, D=1024, H=16, DK=64
// ---------------------------------------------------------------------------
#define Bc   2
#define Sc   2048
#define Dc   1024
#define Hc   16
#define DKc  64
#define Mtot (Bc * Sc)          // 4096 rows for the [*,D] GEMMs
#define EPSc 1e-5f

typedef __bf16 bf16_t;
typedef __attribute__((ext_vector_type(2)))  __bf16 v2bf;
typedef __attribute__((ext_vector_type(16))) __bf16 v16bf;
typedef __attribute__((ext_vector_type(8)))  float  v8f;

union FragAB { v16bf v; unsigned int u[8]; };
union FragC  { v8f   v; float f[8]; };

// ----------------------------- fragment loads ------------------------------
// A-fragment (16x32 bf16, M x K): lane M = l%16, hi = l/16.
// VGPR v packs K = (v/4)*16 + hi*8 + (v%4)*2 , +1   (ISA 7.12.2)
// Per lane this is two contiguous 16B chunks -> 2x b128 loads.
__device__ __forceinline__ FragAB load_fragA_bf16(const bf16_t* rowp, int hi) {
    FragAB f;
#pragma unroll
    for (int v = 0; v < 8; ++v) {
        int k = ((v >> 2) << 4) + (hi << 3) + ((v & 3) << 1);
        f.u[v] = *(const unsigned int*)(rowp + k);
    }
    return f;
}
// B-fragment (32x16, K x N) loaded from a row-major [N, K] array:
// lane N = l%16; VGPR v packs K = (v/4)*16 + hi*8 + (v%4)*2, +1 (same as A)
__device__ __forceinline__ FragAB load_fragB_bf16(const bf16_t* rowp, int hi) {
    FragAB f;
#pragma unroll
    for (int v = 0; v < 8; ++v) {
        int k = ((v >> 2) << 4) + (hi << 3) + ((v & 3) << 1);
        f.u[v] = *(const unsigned int*)(rowp + k);
    }
    return f;
}
__device__ __forceinline__ v8f wmma_bf16(const FragAB& a, const FragAB& b, v8f c) {
    return __builtin_amdgcn_wmma_f32_16x16x32_bf16(
        /*neg_a=*/false, a.v, /*neg_b=*/false, b.v,
        /*c_mod=*/(short)0, c, /*reuse_a=*/false, /*reuse_b=*/false);
}

// ----------------- kernel 0: f32 -> bf16 cast (native cvt) ------------------
__global__ __launch_bounds__(256) void cvt_f32_bf16_kernel(
    const float2* __restrict__ s, v2bf* __restrict__ d, int npairs) {
    for (int i = blockIdx.x * blockDim.x + threadIdx.x; i < npairs;
         i += gridDim.x * blockDim.x) {
        const float2 f = s[i];
        v2bf o;
        o.x = (bf16_t)f.x;
        o.y = (bf16_t)f.y;
        d[i] = o;
    }
}

// ------------- kernel 1: fused QKV projections (x @ W^T + b) ----------------
// grid: (Mtot/128, Dc/64, 3), block 256 (8 waves). Wave w owns rows
// [blk*128 + w*16, +16) x cols [blkY*64, +64) => 4 wmma tiles, K-loop 1024/32.
// All fragment loads preloaded per k-step: one load clause, one wait, 4 wmma.
__global__ __launch_bounds__(256) void qkv_proj_kernel(
    const bf16_t* __restrict__ xq, const bf16_t* __restrict__ xk,
    const bf16_t* __restrict__ xv,
    const bf16_t* __restrict__ wq, const bf16_t* __restrict__ wk,
    const bf16_t* __restrict__ wv,
    const float* __restrict__ bq, const float* __restrict__ bk,
    const float* __restrict__ bv,
    bf16_t* __restrict__ qb,   // [B,H,S,DK]
    bf16_t* __restrict__ kb,   // [B,H,S,DK]
    bf16_t* __restrict__ vt)   // [B,H,DK,S] (transposed for PV B-operand)
{
    const int which = blockIdx.z;
    const bf16_t* X = (which == 0) ? xq : (which == 1) ? xk : xv;
    const bf16_t* W = (which == 0) ? wq : (which == 1) ? wk : wv;
    const float* bias = (which == 0) ? bq : (which == 1) ? bk : bv;

    const int wave = threadIdx.x >> 5;
    const int lane = threadIdx.x & 31;
    const int lr = lane & 15, hi = lane >> 4;

    const int m0 = blockIdx.x * 128 + wave * 16;
    const int n0 = blockIdx.y * 64;

    FragC acc[4];
#pragma unroll
    for (int t = 0; t < 4; ++t)
#pragma unroll
        for (int i = 0; i < 8; ++i) acc[t].f[i] = 0.0f;

    const bf16_t* arow = X + (size_t)(m0 + lr) * Dc;
    const bf16_t* brow0 = W + (size_t)(n0 + lr) * Dc;
    for (int k0 = 0; k0 < Dc; k0 += 32) {
        FragAB a = load_fragA_bf16(arow + k0, hi);
        FragAB b[4];
#pragma unroll
        for (int t = 0; t < 4; ++t)
            b[t] = load_fragB_bf16(brow0 + (size_t)t * 16 * Dc + k0, hi);
#pragma unroll
        for (int t = 0; t < 4; ++t)
            acc[t].v = wmma_bf16(a, b[t], acc[t].v);
    }

#pragma unroll
    for (int t = 0; t < 4; ++t)
#pragma unroll
        for (int r = 0; r < 8; ++r) {
            const int m = m0 + hi * 8 + r;          // row in [0, 4096)
            const int n = n0 + t * 16 + lr;         // col in [0, 1024)
            const bf16_t val = (bf16_t)(acc[t].f[r] + bias[n]);
            const int bb = m >> 11, ss = m & (Sc - 1);
            const int hh = n >> 6,  dk = n & (DKc - 1);
            if (which == 2)
                vt[(((size_t)bb * Hc + hh) * DKc + dk) * Sc + ss] = val;
            else if (which == 1)
                kb[(((size_t)bb * Hc + hh) * Sc + ss) * DKc + dk] = val;
            else
                qb[(((size_t)bb * Hc + hh) * Sc + ss) * DKc + dk] = val;
        }
}

// ---- kernel 2: fused scores = qk^T/8 -> softmax -> attn-out AND O = P@V ----
// grid: (S/16, H, B), block 256 (8 waves).
// LDS: sc[16][2048] f32 (128 KB, dead after phase 2)
//      pbf[16][2048] bf16 (64 KB)
//      opart[8][16*64] f32 (32 KB) ALIASES the start of sc  => 192 KB total.
__global__ __launch_bounds__(256) void attn_fused_kernel(
    const bf16_t* __restrict__ qb, const bf16_t* __restrict__ kb,
    const bf16_t* __restrict__ vt,
    float* __restrict__ attn, bf16_t* __restrict__ ob)
{
    extern __shared__ float smem[];
    float*  sc    = smem;                             // [16][Sc] f32
    bf16_t* pbf   = (bf16_t*)(smem + 16 * Sc);        // [16][Sc] bf16
    float*  opart = smem;                             // aliases sc (phase 3)

    const int wave = threadIdx.x >> 5;
    const int lane = threadIdx.x & 31;
    const int lr = lane & 15, hi = lane >> 4;

    const int bb = blockIdx.z, hh = blockIdx.y;
    const int bh = bb * Hc + hh;
    const bf16_t* qh = qb + (size_t)bh * Sc * DKc;
    const bf16_t* kh = kb + (size_t)bh * Sc * DKc;
    const bf16_t* vh = vt + (size_t)bh * DKc * Sc;
    const int q0 = blockIdx.x * 16;

    // ---- phase 1: scores ----
    const bf16_t* arow = qh + (size_t)(q0 + lr) * DKc;
    FragAB a0 = load_fragA_bf16(arow + 0, hi);
    FragAB a1 = load_fragA_bf16(arow + 32, hi);

    for (int t = 0; t < 16; ++t) {
        const int ki0 = wave * 256 + t * 16;
        const bf16_t* brow = kh + (size_t)(ki0 + lr) * DKc;
        FragAB b0 = load_fragB_bf16(brow + 0, hi);
        FragAB b1 = load_fragB_bf16(brow + 32, hi);
        FragC acc;
#pragma unroll
        for (int i = 0; i < 8; ++i) acc.f[i] = 0.0f;
        acc.v = wmma_bf16(a0, b0, acc.v);
        acc.v = wmma_bf16(a1, b1, acc.v);
#pragma unroll
        for (int r = 0; r < 8; ++r)
            sc[(hi * 8 + r) * Sc + (ki0 + lr)] = acc.f[r] * 0.125f; // /sqrt(64)
    }
    __syncthreads();

    // ---- phase 2: softmax; write f32 attn (HBM) + bf16 P (LDS) ----
    for (int rr = 0; rr < 2; ++rr) {
        const int row = wave * 2 + rr;
        float* srow = sc + (size_t)row * Sc;
        float mx = -3.4e38f;
        for (int c = lane; c < Sc; c += 32) mx = fmaxf(mx, srow[c]);
#pragma unroll
        for (int off = 16; off > 0; off >>= 1)
            mx = fmaxf(mx, __shfl_xor(mx, off, 32));
        float sum = 0.0f;
        for (int c = lane; c < Sc; c += 32) {
            const float e = __expf(srow[c] - mx);
            srow[c] = e;
            sum += e;
        }
#pragma unroll
        for (int off = 16; off > 0; off >>= 1) sum += __shfl_xor(sum, off, 32);
        const float inv = 1.0f / sum;
        float* orow = attn + ((size_t)bh * Sc + q0 + row) * Sc;
        bf16_t* prow = pbf + (size_t)row * Sc;
        for (int c = lane; c < Sc; c += 32) {
            const float p = srow[c] * inv;
            orow[c] = p;
            prow[c] = (bf16_t)p;
        }
    }
    __syncthreads();

    // ---- phase 3: O partial = P[:, w*256 : +256] @ V^T, then reduce ----
    FragC acc[4];
#pragma unroll
    for (int t = 0; t < 4; ++t)
#pragma unroll
        for (int i = 0; i < 8; ++i) acc[t].f[i] = 0.0f;

    const bf16_t* prow = pbf + (size_t)lr * Sc;   // q-row = lr (A-frag M)
    const bf16_t* brow0 = vh + (size_t)lr * Sc;
    for (int kk = 0; kk < 256; kk += 32) {
        const int k0 = wave * 256 + kk;
        FragAB a = load_fragA_bf16(prow + k0, hi);   // ds_load b128 x2
        FragAB b[4];
#pragma unroll
        for (int t = 0; t < 4; ++t)
            b[t] = load_fragB_bf16(brow0 + (size_t)t * 16 * Sc + k0, hi);
#pragma unroll
        for (int t = 0; t < 4; ++t)
            acc[t].v = wmma_bf16(a, b[t], acc[t].v);
    }
    float* op = opart + wave * (16 * DKc);
#pragma unroll
    for (int t = 0; t < 4; ++t)
#pragma unroll
        for (int r = 0; r < 8; ++r)
            op[(hi * 8 + r) * DKc + t * 16 + lr] = acc[t].f[r];
    __syncthreads();

    // 8-way reduction of 16x64 partials, store O as bf16 [B,S,D]
    for (int i = threadIdx.x; i < 16 * DKc; i += 256) {
        float s = 0.0f;
#pragma unroll
        for (int w = 0; w < 8; ++w) s += opart[w * (16 * DKc) + i];
        const int qrow = i >> 6, d = i & (DKc - 1);
        ob[(size_t)(bb * Sc + q0 + qrow) * Dc + hh * DKc + d] = (bf16_t)s;
    }
}

// --------- kernel 3: out-proj + bias + residual (pre-LayerNorm f32) ---------
// grid: (Mtot/128, Dc/64), block 256. A = O (bf16), B = Wo (bf16).
__global__ __launch_bounds__(256) void out_proj_kernel(
    const bf16_t* __restrict__ ob, const bf16_t* __restrict__ wo,
    const float* __restrict__ bo, const float* __restrict__ resid,
    float* __restrict__ tmp)
{
    const int wave = threadIdx.x >> 5;
    const int lane = threadIdx.x & 31;
    const int lr = lane & 15, hi = lane >> 4;

    const int m0 = blockIdx.x * 128 + wave * 16;
    const int n0 = blockIdx.y * 64;

    FragC acc[4];
#pragma unroll
    for (int t = 0; t < 4; ++t)
#pragma unroll
        for (int i = 0; i < 8; ++i) acc[t].f[i] = 0.0f;

    const bf16_t* arow = ob + (size_t)(m0 + lr) * Dc;
    const bf16_t* brow0 = wo + (size_t)(n0 + lr) * Dc;
    for (int k0 = 0; k0 < Dc; k0 += 32) {
        FragAB a = load_fragA_bf16(arow + k0, hi);
        FragAB b[4];
#pragma unroll
        for (int t = 0; t < 4; ++t)
            b[t] = load_fragB_bf16(brow0 + (size_t)t * 16 * Dc + k0, hi);
#pragma unroll
        for (int t = 0; t < 4; ++t)
            acc[t].v = wmma_bf16(a, b[t], acc[t].v);
    }

#pragma unroll
    for (int t = 0; t < 4; ++t)
#pragma unroll
        for (int r = 0; r < 8; ++r) {
            const int m = m0 + hi * 8 + r;
            const int n = n0 + t * 16 + lr;
            tmp[(size_t)m * Dc + n] =
                acc[t].f[r] + bo[n] + resid[(size_t)m * Dc + n];
        }
}

// ----------------------- kernel 4: LayerNorm -------------------------------
__global__ __launch_bounds__(256) void layernorm_kernel(
    const float* __restrict__ x_, const float* __restrict__ gamma,
    const float* __restrict__ beta, float* __restrict__ y)
{
    __shared__ float red1[8];
    __shared__ float red2[8];
    const int row = blockIdx.x;
    const int t = threadIdx.x;
    const int wave = t >> 5, lane = t & 31;
    const float* x = x_ + (size_t)row * Dc;

    float vals[4];
    float lsum = 0.0f;
#pragma unroll
    for (int i = 0; i < 4; ++i) {
        vals[i] = x[t + i * 256];
        lsum += vals[i];
    }
#pragma unroll
    for (int off = 16; off > 0; off >>= 1) lsum += __shfl_xor(lsum, off, 32);
    if (lane == 0) red1[wave] = lsum;
    __syncthreads();
    float tot = 0.0f;
#pragma unroll
    for (int w = 0; w < 8; ++w) tot += red1[w];
    const float mean = tot * (1.0f / Dc);

    float lvar = 0.0f;
#pragma unroll
    for (int i = 0; i < 4; ++i) {
        const float d = vals[i] - mean;
        lvar += d * d;
    }
#pragma unroll
    for (int off = 16; off > 0; off >>= 1) lvar += __shfl_xor(lvar, off, 32);
    if (lane == 0) red2[wave] = lvar;
    __syncthreads();
    float vtot = 0.0f;
#pragma unroll
    for (int w = 0; w < 8; ++w) vtot += red2[w];
    const float inv = rsqrtf(vtot * (1.0f / Dc) + EPSc);

#pragma unroll
    for (int i = 0; i < 4; ++i) {
        const int c = t + i * 256;
        y[(size_t)row * Dc + c] = (vals[i] - mean) * inv * gamma[c] + beta[c];
    }
}

// ---------------------------------------------------------------------------
extern "C" void kernel_launch(void* const* d_in, const int* in_sizes, int n_in,
                              void* d_out, int out_size, void* d_ws, size_t ws_size,
                              hipStream_t stream) {
    const float* Qf  = (const float*)d_in[0];
    const float* Kf  = (const float*)d_in[1];
    const float* Vf  = (const float*)d_in[2];
    const float* WQ  = (const float*)d_in[3];
    const float* bQ  = (const float*)d_in[4];
    const float* WK  = (const float*)d_in[5];
    const float* bK  = (const float*)d_in[6];
    const float* WV  = (const float*)d_in[7];
    const float* bV  = (const float*)d_in[8];
    const float* Wo  = (const float*)d_in[9];
    const float* bo  = (const float*)d_in[10];
    const float* gamma = (const float*)d_in[11];
    const float* beta  = (const float*)d_in[12];

    // workspace carve-up (all 256B aligned)
    char* ws = (char*)d_ws;
    size_t off = 0;
    auto carve = [&](size_t bytes) {
        char* p = ws + off;
        off += (bytes + 255) & ~(size_t)255;
        return p;
    };
    const size_t nDD = (size_t)Dc * Dc;        // 1 Mi elems
    const size_t nXD = (size_t)Mtot * Dc;      // 4 Mi elems
    bf16_t* wq_b = (bf16_t*)carve(nDD * 2);
    bf16_t* wk_b = (bf16_t*)carve(nDD * 2);
    bf16_t* wv_b = (bf16_t*)carve(nDD * 2);
    bf16_t* wo_b = (bf16_t*)carve(nDD * 2);
    bf16_t* xq_b = (bf16_t*)carve(nXD * 2);
    bf16_t* xk_b = (bf16_t*)carve(nXD * 2);
    bf16_t* xv_b = (bf16_t*)carve(nXD * 2);
    bf16_t* qb   = (bf16_t*)carve(nXD * 2);    // [B,H,S,DK]
    bf16_t* kb   = (bf16_t*)carve(nXD * 2);    // [B,H,S,DK]
    bf16_t* vt   = (bf16_t*)carve(nXD * 2);    // [B,H,DK,S]
    bf16_t* ob   = (bf16_t*)carve(nXD * 2);    // [B,S,D]
    float*  tmp  = (float*)carve(nXD * 4);     // pre-LN f32
    (void)ws_size; (void)in_sizes; (void)n_in; (void)out_size;

    float* y    = (float*)d_out;                        // [B,S,D]
    float* attn = (float*)d_out + (size_t)Bc * Sc * Dc; // [B,H,S,S]

    // 0) f32 -> bf16 conversions (native packed cvt)
    cvt_f32_bf16_kernel<<<1024, 256, 0, stream>>>((const float2*)WQ, (v2bf*)wq_b, (int)(nDD / 2));
    cvt_f32_bf16_kernel<<<1024, 256, 0, stream>>>((const float2*)WK, (v2bf*)wk_b, (int)(nDD / 2));
    cvt_f32_bf16_kernel<<<1024, 256, 0, stream>>>((const float2*)WV, (v2bf*)wv_b, (int)(nDD / 2));
    cvt_f32_bf16_kernel<<<1024, 256, 0, stream>>>((const float2*)Wo, (v2bf*)wo_b, (int)(nDD / 2));
    cvt_f32_bf16_kernel<<<2048, 256, 0, stream>>>((const float2*)Qf, (v2bf*)xq_b, (int)(nXD / 2));
    cvt_f32_bf16_kernel<<<2048, 256, 0, stream>>>((const float2*)Kf, (v2bf*)xk_b, (int)(nXD / 2));
    cvt_f32_bf16_kernel<<<2048, 256, 0, stream>>>((const float2*)Vf, (v2bf*)xv_b, (int)(nXD / 2));

    // 1) QKV projections (z selects which)
    qkv_proj_kernel<<<dim3(Mtot / 128, Dc / 64, 3), 256, 0, stream>>>(
        xq_b, xk_b, xv_b, wq_b, wk_b, wv_b, bQ, bK, bV, qb, kb, vt);

    // 2) fused scores + softmax + PV; 192 KB dynamic LDS (opart aliases sc)
    const size_t lds_bytes = 16 * Sc * sizeof(float)      // scores f32
                           + 16 * Sc * sizeof(bf16_t);    // P bf16
    attn_fused_kernel<<<dim3(Sc / 16, Hc, Bc), 256, lds_bytes, stream>>>(
        qb, kb, vt, attn, ob);

    // 3) output projection + bias + residual
    out_proj_kernel<<<dim3(Mtot / 128, Dc / 64), 256, 0, stream>>>(
        ob, wo_b, bo, Qf, tmp);

    // 4) LayerNorm
    layernorm_kernel<<<Mtot, 256, 0, stream>>>(tmp, gamma, beta, y);
}
